// MultiBoxLoss_13340168421885
// MI455X (gfx1250) — compile-verified, matrix operand order checked by reading, not run on hardware
//
#include <hip/hip_runtime.h>
#include <stdint.h>

#define NUM_B 64
#define NUM_P 25600
#define NUM_O 32
#define NBINS 1024
#define BIN_SCALE 64.0f   // bins cover [0, 16) of conf-loss values
#define THRESH 0.35f
#define VAR0 0.1f
#define VAR1 0.2f
#define BLK 256
#define PBLKS (NUM_P / BLK)   // 100, exact

typedef __attribute__((ext_vector_type(2)))  float    v2f;
typedef __attribute__((ext_vector_type(8)))  float    v8f;
typedef __attribute__((ext_vector_type(16))) _Float16 v16h;

// ---------------- CDNA5 async global->LDS copy (guarded) ----------------
#if defined(__gfx1250__) && __has_builtin(__builtin_amdgcn_global_load_async_to_lds_b32)
#define HAVE_ASYNC_LDS 1
typedef __attribute__((address_space(1))) int gint_t;   // global (AS1)
typedef __attribute__((address_space(3))) int lint_t;   // LDS (AS3)
__device__ __forceinline__ void async_copy_f32(const float* gsrc, float* ldst) {
  __builtin_amdgcn_global_load_async_to_lds_b32(
      (gint_t*)(unsigned long long)(const void*)gsrc,
      (lint_t*)(unsigned int)(unsigned long long)(void*)ldst,
      0, 0);
}
__device__ __forceinline__ void async_wait_all() {
#if __has_builtin(__builtin_amdgcn_s_wait_asynccnt)
  __builtin_amdgcn_s_wait_asynccnt(0);
#else
  asm volatile("s_wait_asynccnt 0" ::: "memory");
#endif
}
#else
#define HAVE_ASYNC_LDS 0
#endif

// ---------------- WMMA ones-matmul wave reduction (exact fp32) ----------------
// D = A(16x4) * ones(4x16): D[m][n] = sum_k A[m][k]. Lane L<16 contributes at
// A[L][0], lane L>=16 at A[L-16][2]. Row-sum extraction: per-lane s = sum of 8
// C VGPRs, wave total = s(lane 0) + s(lane 16).
__device__ __forceinline__ float wave_reduce_sum(float v) {
#if __has_builtin(__builtin_amdgcn_wmma_f32_16x16x4_f32)
  v2f a  = {v, 0.0f};
  v2f bm = {1.0f, 1.0f};
  v8f c  = {};
  c = __builtin_amdgcn_wmma_f32_16x16x4_f32(false, a, false, bm, (short)0, c,
                                            false, false);
  float s = ((c[0] + c[1]) + (c[2] + c[3])) + ((c[4] + c[5]) + (c[6] + c[7]));
  return __shfl(s, 0, 32) + __shfl(s, 16, 32);
#elif __has_builtin(__builtin_amdgcn_wmma_f32_16x16x32_f16)
  v16h a, bm;
#pragma unroll
  for (int i = 0; i < 16; ++i) { a[i] = (_Float16)0.0f; bm[i] = (_Float16)1.0f; }
  a[0] = (_Float16)v;
  v8f c = {};
  c = __builtin_amdgcn_wmma_f32_16x16x32_f16(false, a, false, bm, (short)0, c,
                                             false, false);
  float s = ((c[0] + c[1]) + (c[2] + c[3])) + ((c[4] + c[5]) + (c[6] + c[7]));
  return __shfl(s, 0, 32) + __shfl(s, 16, 32);
#else
#pragma unroll
  for (int off = 16; off; off >>= 1) v += __shfl_xor(v, off, 32);
  return v;
#endif
}

// ---------------- kernels ----------------
__global__ void mbl_init_kernel(const float* __restrict__ priors,
                                float* __restrict__ pf,
                                unsigned long long* __restrict__ pack,
                                int* __restrict__ hist, int* __restrict__ pos_cnt,
                                int* __restrict__ thrb, int* __restrict__ thrr,
                                int* __restrict__ tie, float* __restrict__ acc,
                                int* __restrict__ total_pos) {
  int i = blockIdx.x * BLK + threadIdx.x;  // 65536 threads total
  if (i < NUM_P) {
    float4 p = ((const float4*)priors)[i];
    float4 q;
    q.x = p.x - 0.5f * p.z; q.y = p.y - 0.5f * p.w;
    q.z = p.x + 0.5f * p.z; q.w = p.y + 0.5f * p.w;
    ((float4*)pf)[i] = q;
  }
  if (i < NUM_B * NBINS) hist[i] = 0;
  if (i < NUM_B * NUM_O) pack[i] = 0ull;
  if (i < NUM_B) { pos_cnt[i] = 0; thrb[i] = 0; thrr[i] = 0; tie[i] = 0; }
  if (i == 0) { acc[0] = 0.0f; acc[1] = 0.0f; *total_pos = 0; }
}

__global__ void mbl_match_kernel(const float* __restrict__ targets,
                                 const float* __restrict__ pf,
                                 float* __restrict__ bov, int* __restrict__ bid,
                                 unsigned long long* __restrict__ pack) {
  const int b = blockIdx.y;
  const int j = blockIdx.x * BLK + threadIdx.x;
  const int t = threadIdx.x;
  __shared__ float s_t[NUM_O * 5];
  __shared__ float s_area[NUM_O];
  __shared__ unsigned long long s_best[NUM_O];

  const float* tg = targets + b * NUM_O * 5;
#if HAVE_ASYNC_LDS
  if (t < NUM_O * 5) async_copy_f32(tg + t, &s_t[t]);
  async_wait_all();
#else
  if (t < NUM_O * 5) s_t[t] = tg[t];
#endif
  if (t < NUM_O) s_best[t] = 0ull;
  __syncthreads();
  if (t < NUM_O) {
    float w = s_t[t * 5 + 2] - s_t[t * 5 + 0];
    float h = s_t[t * 5 + 3] - s_t[t * 5 + 1];
    s_area[t] = w * h;
  }
  __syncthreads();

  float4 p = ((const float4*)pf)[j];
  float pa = (p.z - p.x) * (p.w - p.y);
  float best = -1.0f;
  int bi = 0;
#pragma unroll 4
  for (int o = 0; o < NUM_O; ++o) {
    const float* tb = &s_t[o * 5];
    float lx = fmaxf(tb[0], p.x), ly = fmaxf(tb[1], p.y);
    float rx = fminf(tb[2], p.z), ry = fminf(tb[3], p.w);
    float iw = fmaxf(rx - lx, 0.0f), ih = fmaxf(ry - ly, 0.0f);
    float inter = iw * ih;
    float iou = inter / (s_area[o] + pa - inter);
    if (iou > best) { best = iou; bi = o; }
    unsigned long long pk =
        ((unsigned long long)__float_as_uint(iou) << 32) | (unsigned)j;
    if (pk > s_best[o]) atomicMax(&s_best[o], pk);  // monotone filter, DS_MAX_U64
  }
  bov[b * NUM_P + j] = best;
  bid[b * NUM_P + j] = bi;
  __syncthreads();
  if (t < NUM_O) atomicMax(&pack[b * NUM_O + t], s_best[t]);
}

__global__ void mbl_force_kernel(const unsigned long long* __restrict__ pack,
                                 float* __restrict__ bov, int* __restrict__ bid) {
  const int b = blockIdx.x;
  if (threadIdx.x == 0) {
    for (int o = 0; o < NUM_O; ++o) {  // serial: last writer wins, scatter order
      unsigned j = (unsigned)(pack[b * NUM_O + o] & 0xffffffffu);
      bov[b * NUM_P + j] = 2.0f;
      bid[b * NUM_P + j] = o;
    }
  }
}

__global__ void mbl_loss_prep_kernel(const float* __restrict__ loc,
                                     const float* __restrict__ conf,
                                     const float* __restrict__ priors,
                                     const float* __restrict__ targets,
                                     const float* __restrict__ bov,
                                     const int* __restrict__ bid,
                                     float* __restrict__ rankv,
                                     int* __restrict__ hist,
                                     int* __restrict__ pos_cnt,
                                     float* __restrict__ acc,
                                     int* __restrict__ total_pos) {
  const int b = blockIdx.y;
  const int j = blockIdx.x * BLK + threadIdx.x;
  const int t = threadIdx.x;
  __shared__ float s_t[NUM_O * 5];
  __shared__ int s_hist[NBINS];
  __shared__ float s_red[2];
  __shared__ int s_pos;

#if HAVE_ASYNC_LDS
  if (t < NUM_O * 5) async_copy_f32(targets + b * NUM_O * 5 + t, &s_t[t]);
  async_wait_all();
#else
  if (t < NUM_O * 5) s_t[t] = targets[b * NUM_O * 5 + t];
#endif
  for (int i = t; i < NBINS; i += BLK) s_hist[i] = 0;
  if (t == 0) { s_red[0] = 0.0f; s_red[1] = 0.0f; s_pos = 0; }
  __syncthreads();

  const int gix = b * NUM_P + j;
  __builtin_prefetch(&loc[4 * (gix + BLK)], 0, 0);   // global_prefetch_b8
  __builtin_prefetch(&conf[2 * (gix + BLK)], 0, 0);

  float ov = bov[gix];
  int ti = bid[gix];
  const float* tb = &s_t[ti * 5];
  float lab = tb[4];
  int cls = (ov < THRESH) ? 0 : min((int)lab + 1, 1);
  bool pos = cls > 0;

  // classification loss: logsumexp - gathered
  float c0 = conf[gix * 2 + 0], c1 = conf[gix * 2 + 1];
  float m = fmaxf(c0, c1);
  float lse = m + logf(__expf(c0 - m) + __expf(c1 - m));
  float lc = lse - ((cls == 0) ? c0 : c1);
  float rv = pos ? 0.0f : lc;
  rankv[gix] = rv;
  int bin = (int)(rv * BIN_SCALE);
  bin = max(0, min(bin, NBINS - 1));
  atomicAdd(&s_hist[bin], 1);

  // localization: encode matched box vs prior, smooth-L1 where pos
  float4 pr = ((const float4*)priors)[j];
  float mcx = 0.5f * (tb[0] + tb[2]), mcy = 0.5f * (tb[1] + tb[3]);
  float mw = tb[2] - tb[0], mh = tb[3] - tb[1];
  float lt0 = (mcx - pr.x) / (VAR0 * pr.z);
  float lt1 = (mcy - pr.y) / (VAR0 * pr.w);
  float lt2 = logf(mw / pr.z) / VAR1;
  float lt3 = logf(mh / pr.w) / VAR1;
  float4 ld = ((const float4*)loc)[gix];
  float ll = 0.0f;
  if (pos) {
    float d[4] = {ld.x - lt0, ld.y - lt1, ld.z - lt2, ld.w - lt3};
#pragma unroll
    for (int k = 0; k < 4; ++k) {
      float ad = fabsf(d[k]);
      ll += (ad < 1.0f) ? 0.5f * ad * ad : ad - 0.5f;
    }
  }
  float lcp = pos ? lc : 0.0f;

  unsigned long long bal = __ballot(pos);
  float w0 = wave_reduce_sum(ll);    // v_wmma_f32_16x16x4_f32
  float w1 = wave_reduce_sum(lcp);
  if ((t & 31) == 0) {
    atomicAdd(&s_red[0], w0);
    atomicAdd(&s_red[1], w1);
    atomicAdd(&s_pos, (int)__popcll(bal));
  }
  __syncthreads();
  for (int i = t; i < NBINS; i += BLK) {
    int c = s_hist[i];
    if (c) atomicAdd(&hist[b * NBINS + i], c);
  }
  if (t == 0) {
    atomicAdd(&acc[0], s_red[0]);
    atomicAdd(&acc[1], s_red[1]);
    atomicAdd(&pos_cnt[b], s_pos);
    atomicAdd(total_pos, s_pos);
  }
}

__global__ void mbl_select_kernel(const int* __restrict__ hist,
                                  const int* __restrict__ pos_cnt,
                                  int* __restrict__ thrb, int* __restrict__ thrr) {
  int b = threadIdx.x;
  if (b < NUM_B) {
    long long nn = 3LL * pos_cnt[b];
    if (nn > NUM_P - 1) nn = NUM_P - 1;
    int cum = 0, tbn = NBINS, rem = 0;
    for (int bin = NBINS - 1; bin >= 0; --bin) {
      int c = hist[b * NBINS + bin];
      if ((long long)cum + c >= nn) { tbn = bin; rem = (int)(nn - cum); break; }
      cum += c;
    }
    thrb[b] = tbn;
    thrr[b] = rem;
  }
}

__global__ void mbl_neg_sum_kernel(const float* __restrict__ rankv,
                                   const int* __restrict__ thrb,
                                   const int* __restrict__ thrr,
                                   int* __restrict__ tie,
                                   float* __restrict__ acc) {
  const int b = blockIdx.y;
  const int j = blockIdx.x * BLK + threadIdx.x;
  const int gix = b * NUM_P + j;
  __builtin_prefetch(&rankv[gix + BLK], 0, 0);
  float v = rankv[gix];
  int bin = max(0, min((int)(v * BIN_SCALE), NBINS - 1));
  int tbn = thrb[b];
  float contrib = 0.0f;
  if (bin > tbn) {
    contrib = v;
  } else if (bin == tbn) {
    int tk = atomicAdd(&tie[b], 1);  // within-bin quota (tie handling)
    if (tk < thrr[b]) contrib = v;
  }
  float s = wave_reduce_sum(contrib);  // v_wmma_f32_16x16x4_f32
  __shared__ float s_acc;
  if (threadIdx.x == 0) s_acc = 0.0f;
  __syncthreads();
  if ((threadIdx.x & 31) == 0) atomicAdd(&s_acc, s);
  __syncthreads();
  if (threadIdx.x == 0) atomicAdd(&acc[1], s_acc);
}

__global__ void mbl_finalize_kernel(const float* __restrict__ acc,
                                    const int* __restrict__ total_pos,
                                    float* __restrict__ out) {
  if (threadIdx.x == 0 && blockIdx.x == 0) {
    float N = fmaxf((float)*total_pos, 1.0f);
    out[0] = acc[0] / N;
    out[1] = acc[1] / N;
  }
}

// ---------------- launch ----------------
extern "C" void kernel_launch(void* const* d_in, const int* in_sizes, int n_in,
                              void* d_out, int out_size, void* d_ws, size_t ws_size,
                              hipStream_t stream) {
  (void)in_sizes; (void)n_in; (void)out_size; (void)ws_size;
  const float* loc     = (const float*)d_in[0];  // B*P*4
  const float* conf    = (const float*)d_in[1];  // B*P*2
  const float* priors  = (const float*)d_in[2];  // P*4
  const float* targets = (const float*)d_in[3];  // B*O*5
  float* out = (float*)d_out;                    // 2 floats

  char* w = (char*)d_ws;
  unsigned long long* pack = (unsigned long long*)(w + 0);        // B*O u64
  float* pf        = (float*)(w + 16384);                         // P*4 f
  float* bov       = (float*)(w + 425984);                        // B*P f
  int*   bid       = (int*)  (w + 6979584);                       // B*P i
  float* rankv     = (float*)(w + 13533184);                      // B*P f
  int*   hist      = (int*)  (w + 20086784);                      // B*NBINS i
  int*   pos_cnt   = (int*)  (w + 20348928);                      // B i
  int*   thrb      = (int*)  (w + 20349184);                      // B i
  int*   thrr      = (int*)  (w + 20349440);                      // B i
  int*   tie       = (int*)  (w + 20349696);                      // B i
  float* acc       = (float*)(w + 20349952);                      // [loss_l, loss_c]
  int*   total_pos = (int*)  (w + 20349960);

  dim3 gbp(PBLKS, NUM_B);
  mbl_init_kernel<<<256, BLK, 0, stream>>>(priors, pf, pack, hist, pos_cnt,
                                           thrb, thrr, tie, acc, total_pos);
  mbl_match_kernel<<<gbp, BLK, 0, stream>>>(targets, pf, bov, bid, pack);
  mbl_force_kernel<<<NUM_B, 32, 0, stream>>>(pack, bov, bid);
  mbl_loss_prep_kernel<<<gbp, BLK, 0, stream>>>(loc, conf, priors, targets, bov,
                                                bid, rankv, hist, pos_cnt, acc,
                                                total_pos);
  mbl_select_kernel<<<1, 64, 0, stream>>>(hist, pos_cnt, thrb, thrr);
  mbl_neg_sum_kernel<<<gbp, BLK, 0, stream>>>(rankv, thrb, thrr, tie, acc);
  mbl_finalize_kernel<<<1, 32, 0, stream>>>(acc, total_pos, out);
}
